// NLITreeLSTM_26087631356096
// MI455X (gfx1250) — matrix operand order repaired and checked by reading, not compiled, not run on hardware
//
#include <hip/hip_runtime.h>
#include <hip/hip_bf16.h>

// ---------- types ----------
typedef __attribute__((ext_vector_type(8)))  __bf16 bf16x8;
typedef __attribute__((ext_vector_type(16))) __bf16 v16bf;
typedef __attribute__((ext_vector_type(8)))  float  v8f;

#define BSZ   512
#define SEQ   128
#define DIM   300
#define HID   300
#define HP    304      // H padded to 16
#define KP    320      // K padded to 32 (for D and H as contraction dims)
#define NP    1216     // 4*HP
#define KP2   1216     // classifier contraction (4*HP), multiple of 32
#define NCLS  3

static __device__ __forceinline__ __bf16 f2bf(float f) {
    unsigned u = __builtin_bit_cast(unsigned, f);
    u += 0x7FFFu + ((u >> 16) & 1u);              // round-to-nearest-even
    unsigned short s = (unsigned short)(u >> 16);
    return __builtin_bit_cast(__bf16, s);
}
static __device__ __forceinline__ float bf2f(__bf16 b) {
    unsigned short s = __builtin_bit_cast(unsigned short, b);
    unsigned u = ((unsigned)s) << 16;
    return __builtin_bit_cast(float, u);
}
static __device__ __forceinline__ float sigm(float x) {
    return 1.0f / (1.0f + __expf(-x));
}

// ---------- prep kernels ----------
__global__ void convert_x_kernel(const float* __restrict__ lx,
                                 const float* __restrict__ rx,
                                 __bf16* __restrict__ xb) {
    size_t id = (size_t)blockIdx.x * blockDim.x + threadIdx.x;
    const size_t total = (size_t)2 * BSZ * SEQ * KP;
    if (id >= total) return;
    int k    = (int)(id % KP);
    size_t r = id / KP;              // tree*BSZ*SEQ + b*SEQ + s
    int s    = (int)(r % SEQ);
    size_t r2 = r / SEQ;
    int b    = (int)(r2 % BSZ);
    int tree = (int)(r2 / BSZ);
    float v = 0.0f;
    if (k < DIM) {
        const float* src = tree ? rx : lx;
        v = src[((size_t)b * SEQ + s) * DIM + k];
    }
    xb[id] = f2bf(v);
}

__global__ void prep_w_kernel(const float* __restrict__ Wioux,
                              const float* __restrict__ Wiouh,
                              const float* __restrict__ Wfx,
                              const float* __restrict__ Wfh,
                              const float* __restrict__ bioux,
                              const float* __restrict__ biouh,
                              const float* __restrict__ bfx,
                              const float* __restrict__ bfh,
                              __bf16* __restrict__ Wx,
                              __bf16* __restrict__ Wh,
                              float* __restrict__ bias) {
    int id = blockIdx.x * blockDim.x + threadIdx.x;
    if (id >= NP * KP) return;
    int n = id / KP, k = id % KP;
    int g = n / HP, j = n % HP;      // gate 0=i 1=o 2=u 3=f
    float vx = 0.0f, vh = 0.0f;
    if (j < HID && k < DIM) {
        if (g < 3) {
            vx = Wioux[((size_t)(g * HID + j)) * DIM + k];
            vh = Wiouh[((size_t)(g * HID + j)) * HID + k];
        } else {
            vx = Wfx[(size_t)j * DIM + k];
            vh = Wfh[(size_t)j * HID + k];
        }
    }
    Wx[id] = f2bf(vx);
    Wh[id] = f2bf(vh);
    if (k == 0) {
        float bv = 0.0f;
        if (j < HID) bv = (g < 3) ? (bioux[g * HID + j] + biouh[g * HID + j])
                                  : (bfx[j] + bfh[j]);
        bias[n] = bv;
    }
}

__global__ void prep_w1_kernel(const float* __restrict__ W1,
                               const float* __restrict__ b1,
                               __bf16* __restrict__ W1p,
                               float* __restrict__ b1p) {
    int id = blockIdx.x * blockDim.x + threadIdx.x;
    if (id >= HP * KP2) return;
    int n = id / KP2, q = id % KP2;
    int s = q / HP, j = q % HP;
    float v = 0.0f;
    if (n < HID && j < HID) v = W1[(size_t)n * (4 * HID) + s * HID + j];
    W1p[id] = f2bf(v);
    if (q == 0) b1p[n] = (n < HID) ? b1[n] : 0.0f;
}

// ---------- recurrent step: fused dual-GEMM (x & h) + LSTM cell ----------
__global__ __launch_bounds__(128)
void lstm_step_kernel(const __bf16* __restrict__ xb,
                      const __bf16* __restrict__ Wx,
                      const __bf16* __restrict__ Wh,
                      const float*  __restrict__ bias,
                      const __bf16* __restrict__ hprev,
                      __bf16* __restrict__ hnext,
                      float* __restrict__ cbuf,
                      int t) {
    const int lane = threadIdx.x & 31;
    const int gate = threadIdx.x >> 5;          // wave id == gate id
    const int j0   = blockIdx.x * 16;           // hidden-unit tile
    const int m0   = blockIdx.y * 16;           // batch tile
    const int tree = blockIdx.z;

    // A operand: lane<16 holds row, K = k0+0..7 & k0+16..23; lane>=16: +8
    const int arow  = m0 + (lane & 15);
    const int koffA = (lane >> 4) * 8;
    const __bf16* xrow = xb + ((size_t)tree * BSZ * SEQ + (size_t)arow * SEQ + t) * KP;
    const __bf16* hrow = hprev + ((size_t)tree * BSZ + arow) * KP;
    // B operand (weights as [N][KP]): lane<16 col n, K = k0+0..7 & k0+16..23; lane>=16: +8
    const int ncol  = gate * HP + j0 + (lane & 15);
    const int koffB = (lane >> 4) * 8;
    const __bf16* wxrow = Wx + (size_t)ncol * KP;
    const __bf16* whrow = Wh + (size_t)ncol * KP;

    __builtin_prefetch(whrow, 0, 3);

    v8f acc = {};
#pragma unroll
    for (int kt = 0; kt < KP / 32; ++kt) {
        const int k0 = kt * 32;
        union { v16bf v; bf16x8 h[2]; } a, b;
        a.h[0] = *(const bf16x8*)(xrow + k0 + koffA);
        a.h[1] = *(const bf16x8*)(xrow + k0 + koffA + 16);
        b.h[0] = *(const bf16x8*)(wxrow + k0 + koffB);
        b.h[1] = *(const bf16x8*)(wxrow + k0 + koffB + 16);
        acc = __builtin_amdgcn_wmma_f32_16x16x32_bf16(false, a.v, false, b.v,
                                                      (short)0, acc, false, false);
    }
#pragma unroll
    for (int kt = 0; kt < KP / 32; ++kt) {
        const int k0 = kt * 32;
        union { v16bf v; bf16x8 h[2]; } a, b;
        a.h[0] = *(const bf16x8*)(hrow + k0 + koffA);
        a.h[1] = *(const bf16x8*)(hrow + k0 + koffA + 16);
        b.h[0] = *(const bf16x8*)(whrow + k0 + koffB);
        b.h[1] = *(const bf16x8*)(whrow + k0 + koffB + 16);
        acc = __builtin_amdgcn_wmma_f32_16x16x32_bf16(false, a.v, false, b.v,
                                                      (short)0, acc, false, false);
    }

    // Exchange gate tiles: C layout VGPR r -> (M = r | r+8, N = lane&15)
    __shared__ float g[4 * 256];
#pragma unroll
    for (int r = 0; r < 8; ++r) g[gate * 256 + r * 32 + lane] = acc[r];
    __syncthreads();

    // Cell update: 256 (m, j) elements across 128 threads
    for (int e = threadIdx.x; e < 256; e += 128) {
        const int m = e >> 4, j = e & 15;
        const int loc = ((m & 7) << 5) + ((m >> 3) << 4) + j;
        const float gi = g[0 * 256 + loc] + bias[0 * HP + j0 + j];
        const float go = g[1 * 256 + loc] + bias[1 * HP + j0 + j];
        const float gu = g[2 * 256 + loc] + bias[2 * HP + j0 + j];
        const float gf = g[3 * 256 + loc] + bias[3 * HP + j0 + j];
        const float iv = sigm(gi), ov = sigm(go), fv = sigm(gf);
        const float uv = tanhf(gu);
        const size_t cidx = ((size_t)tree * BSZ + (m0 + m)) * HP + (j0 + j);
        const float cp = cbuf[cidx];
        const float cn = iv * uv + fv * cp;
        cbuf[cidx] = cn;
        const float hv = ov * tanhf(cn);
        hnext[((size_t)tree * BSZ + (m0 + m)) * KP + (j0 + j)] = f2bf(hv);
    }
}

// ---------- classifier ----------
__global__ void vec_build_kernel(const __bf16* __restrict__ hfin,
                                 __bf16* __restrict__ vec) {
    int id = blockIdx.x * blockDim.x + threadIdx.x;
    if (id >= BSZ * KP2) return;
    int b = id / KP2, q = id % KP2;
    int s = q / HP, j = q % HP;
    float lh = bf2f(hfin[((size_t)0 * BSZ + b) * KP + j]);   // pads are exactly 0
    float rh = bf2f(hfin[((size_t)1 * BSZ + b) * KP + j]);
    float v = (s == 0) ? lh : (s == 1) ? rh : (s == 2) ? lh * rh : fabsf(lh - rh);
    vec[id] = f2bf(v);
}

__global__ __launch_bounds__(32)
void gemm_cls_kernel(const __bf16* __restrict__ vec,
                     const __bf16* __restrict__ W1p,
                     const float*  __restrict__ b1p,
                     float* __restrict__ out1) {
    const int lane = threadIdx.x & 31;
    const int n0 = blockIdx.x * 16, m0 = blockIdx.y * 16;
    const int arow  = m0 + (lane & 15);
    const int koffA = (lane >> 4) * 8;
    const int koffB = (lane >> 4) * 8;
    const __bf16* arowp = vec + (size_t)arow * KP2;
    const __bf16* brow  = W1p + (size_t)(n0 + (lane & 15)) * KP2;
    v8f acc = {};
#pragma unroll
    for (int kt = 0; kt < KP2 / 32; ++kt) {
        const int k0 = kt * 32;
        union { v16bf v; bf16x8 h[2]; } a, b;
        a.h[0] = *(const bf16x8*)(arowp + k0 + koffA);
        a.h[1] = *(const bf16x8*)(arowp + k0 + koffA + 16);
        b.h[0] = *(const bf16x8*)(brow + k0 + koffB);
        b.h[1] = *(const bf16x8*)(brow + k0 + koffB + 16);
        acc = __builtin_amdgcn_wmma_f32_16x16x32_bf16(false, a.v, false, b.v,
                                                      (short)0, acc, false, false);
    }
    const int ncol = n0 + (lane & 15);
    const float bv = b1p[ncol];
#pragma unroll
    for (int r = 0; r < 8; ++r) {
        const int m = m0 + r + ((lane >> 4) << 3);
        out1[(size_t)m * HP + ncol] = fmaxf(acc[r] + bv, 0.0f);
    }
}

__global__ __launch_bounds__(32)
void logits_kernel(const float* __restrict__ out1,
                   const float* __restrict__ W2,
                   const float* __restrict__ b2,
                   float* __restrict__ out) {
    const int b = blockIdx.x, lane = threadIdx.x;
    float a0 = 0.f, a1 = 0.f, a2 = 0.f;
    for (int j = lane; j < HID; j += 32) {
        const float v = out1[(size_t)b * HP + j];
        a0 += v * W2[0 * HID + j];
        a1 += v * W2[1 * HID + j];
        a2 += v * W2[2 * HID + j];
    }
#pragma unroll
    for (int off = 16; off > 0; off >>= 1) {
        a0 += __shfl_xor(a0, off);
        a1 += __shfl_xor(a1, off);
        a2 += __shfl_xor(a2, off);
    }
    const float l0 = a0 + b2[0], l1 = a1 + b2[1], l2 = a2 + b2[2];
    const float mx = fmaxf(l0, fmaxf(l1, l2));
    const float ls = logf(expf(l0 - mx) + expf(l1 - mx) + expf(l2 - mx));
    if (lane == 0) {
        out[b * NCLS + 0] = l0 - mx - ls;
        out[b * NCLS + 1] = l1 - mx - ls;
        out[b * NCLS + 2] = l2 - mx - ls;
    }
}

// ---------- host ----------
static inline size_t alignup(size_t x) { return (x + 255) & ~(size_t)255; }

extern "C" void kernel_launch(void* const* d_in, const int* in_sizes, int n_in,
                              void* d_out, int out_size, void* d_ws, size_t ws_size,
                              hipStream_t stream) {
    (void)in_sizes; (void)n_in; (void)out_size; (void)ws_size;
    const float* lx    = (const float*)d_in[0];
    const float* rx    = (const float*)d_in[1];
    const float* Wioux = (const float*)d_in[2];
    const float* bioux = (const float*)d_in[3];
    const float* Wiouh = (const float*)d_in[4];
    const float* biouh = (const float*)d_in[5];
    const float* Wfx   = (const float*)d_in[6];
    const float* bfx   = (const float*)d_in[7];
    const float* Wfh   = (const float*)d_in[8];
    const float* bfh   = (const float*)d_in[9];
    const float* W1    = (const float*)d_in[10];
    const float* b1    = (const float*)d_in[11];
    const float* W2    = (const float*)d_in[12];
    const float* b2    = (const float*)d_in[13];

    char* ws = (char*)d_ws;
    size_t off = 0;
    __bf16* xb  = (__bf16*)(ws + off); off += alignup((size_t)2 * BSZ * SEQ * KP * 2);
    __bf16* Wx  = (__bf16*)(ws + off); off += alignup((size_t)NP * KP * 2);
    __bf16* Wh  = (__bf16*)(ws + off); off += alignup((size_t)NP * KP * 2);
    float*  bia = (float*)(ws + off);  off += alignup((size_t)NP * 4);
    __bf16* h0  = (__bf16*)(ws + off); off += alignup((size_t)2 * BSZ * KP * 2);
    __bf16* h1  = (__bf16*)(ws + off); off += alignup((size_t)2 * BSZ * KP * 2);
    float*  cb  = (float*)(ws + off);  off += alignup((size_t)2 * BSZ * HP * 4);
    __bf16* vec = (__bf16*)(ws + off); off += alignup((size_t)BSZ * KP2 * 2);
    __bf16* W1p = (__bf16*)(ws + off); off += alignup((size_t)HP * KP2 * 2);
    float*  b1p = (float*)(ws + off);  off += alignup((size_t)HP * 4);
    float*  o1  = (float*)(ws + off);  off += alignup((size_t)BSZ * HP * 4);

    hipMemsetAsync(h0, 0, (size_t)2 * BSZ * KP * 2, stream);
    hipMemsetAsync(h1, 0, (size_t)2 * BSZ * KP * 2, stream);
    hipMemsetAsync(cb, 0, (size_t)2 * BSZ * HP * 4, stream);

    {
        const size_t total = (size_t)2 * BSZ * SEQ * KP;
        convert_x_kernel<<<(unsigned)((total + 255) / 256), 256, 0, stream>>>(lx, rx, xb);
    }
    prep_w_kernel<<<(NP * KP + 255) / 256, 256, 0, stream>>>(
        Wioux, Wiouh, Wfx, Wfh, bioux, biouh, bfx, bfh, Wx, Wh, bia);
    prep_w1_kernel<<<(HP * KP2 + 255) / 256, 256, 0, stream>>>(W1, b1, W1p, b1p);

    __bf16* hbuf[2] = { h0, h1 };
    for (int step = 0; step < SEQ; ++step) {
        const int t = SEQ - 1 - step;     // leaf (t=S-1) first, root (t=0) last
        lstm_step_kernel<<<dim3(HP / 16, BSZ / 16, 2), 128, 0, stream>>>(
            xb, Wx, Wh, bia, hbuf[step & 1], hbuf[(step + 1) & 1], cb, t);
    }
    // after SEQ (even) steps, final h is in hbuf[0]

    vec_build_kernel<<<(BSZ * KP2 + 255) / 256, 256, 0, stream>>>(h0, vec);
    gemm_cls_kernel<<<dim3(HP / 16, BSZ / 16), 32, 0, stream>>>(vec, W1p, b1p, o1);
    logits_kernel<<<BSZ, 32, 0, stream>>>(o1, W2, b2, (float*)d_out);
}
